// Hierarchical_Perceiver_35914516529672
// MI455X (gfx1250) — matrix operand order repaired and checked by reading, not compiled, not run on hardware
//
#include <hip/hip_runtime.h>
#include <math.h>

// ---------------- types / helpers ----------------
typedef __attribute__((ext_vector_type(16))) __bf16 v16bf;
typedef __attribute__((ext_vector_type(8)))  float v8f;
typedef __attribute__((ext_vector_type(16))) unsigned short v16us;

__device__ __forceinline__ unsigned short f2bf(float f) {
  unsigned u = __builtin_bit_cast(unsigned, f);
  return (unsigned short)((u + 0x7FFFu + ((u >> 16) & 1u)) >> 16);  // RNE
}

static inline int imin_(int a, int b) { return a < b ? a : b; }

// ---------------- weight transpose+convert: W[ws,K,N](f32) -> Wt[ws,N,K](bf16) ----------------
__global__ void wt_kernel(const float* __restrict__ W, unsigned short* __restrict__ Wt,
                          int K, int N)
{
  __shared__ float tile[16][17];
  const int ws = blockIdx.z;
  const int n0 = blockIdx.x * 16, k0 = blockIdx.y * 16;
  const int tx = threadIdx.x, ty = threadIdx.y;
  const float* Wb = W + (size_t)ws * K * N;
  tile[ty][tx] = Wb[(size_t)(k0 + ty) * N + n0 + tx];
  __syncthreads();
  Wt[(size_t)ws * N * K + (size_t)(n0 + ty) * K + k0 + tx] = f2bf(tile[tx][ty]);
}

// ---------------- GEMM: C[s,M,N] = A[s/Ap,M,K](bf16) @ Wt[s/Wp,N,K](bf16,N-major) + bias ----------
// One wave computes a 64x16 output strip: 4 WMMAs per K-step sharing one B fragment.
// Both operands are contiguous 2x16B chunks per lane -> global_load_b128 pairs.
__global__ __launch_bounds__(32) void gemm_bf16_kernel(
    const unsigned short* __restrict__ A, const unsigned short* __restrict__ Wt,
    const float* __restrict__ bias, float* __restrict__ C,
    int M, int N, int K, int A_period, int W_period)
{
  const int lane = threadIdx.x;
  const int n0 = blockIdx.x * 16;
  const int m0 = blockIdx.y * 64;
  const int s  = blockIdx.z;
  const int ws = s / W_period;
  const unsigned short* Ab = A + (size_t)(s / A_period) * M * K;
  const unsigned short* Wb = Wt + (size_t)ws * N * K;
  const int half = lane >> 4;
  const int mr = lane & 15;   // row-in-tile for A, col-in-tile for B/C
  v8f acc[4] = {};
  for (int k0 = 0; k0 < K; k0 += 32) {
    // B fragment: element e is K = k0 + half*16 + e at column n0+mr -> contiguous in Wt
    const unsigned short* wrow = Wb + (size_t)(n0 + mr) * K + k0 + half * 16;
    v16us bu;
#pragma unroll
    for (int j = 0; j < 8; ++j) {
      unsigned pair = *(const unsigned*)(wrow + 2 * j);
      bu[2 * j]     = (unsigned short)(pair & 0xFFFFu);
      bu[2 * j + 1] = (unsigned short)(pair >> 16);
    }
    v16bf b = __builtin_bit_cast(v16bf, bu);
#pragma unroll
    for (int t = 0; t < 4; ++t) {
      const unsigned short* arow = Ab + (size_t)(m0 + t * 16 + mr) * K;
      v16us au;
#pragma unroll
      for (int j = 0; j < 8; ++j) {
        int kb = k0 + half * 16 + 2 * j;
        unsigned pair = *(const unsigned*)(arow + kb);
        au[2 * j]     = (unsigned short)(pair & 0xFFFFu);
        au[2 * j + 1] = (unsigned short)(pair >> 16);
      }
      v16bf a = __builtin_bit_cast(v16bf, au);
      acc[t] = __builtin_amdgcn_wmma_f32_16x16x32_bf16(false, a, false, b,
                                                       (short)0, acc[t], false, false);
    }
  }
  const float bv = bias ? bias[(size_t)ws * N + n0 + mr] : 0.0f;
  float* Cb = C + (size_t)s * M * N;
#pragma unroll
  for (int t = 0; t < 4; ++t)
#pragma unroll
    for (int r = 0; r < 8; ++r)
      Cb[(size_t)(m0 + t * 16 + half * 8 + r) * N + n0 + mr] = acc[t][r] + bv;
}

// ---------------- S[z,Lq,Lkv] = scale * Q @ K^T  (per (gb,head) slice t = t_base+z) ----------------
__global__ __launch_bounds__(32) void attn_scores_kernel(
    const float* __restrict__ Q, const float* __restrict__ KV, float* __restrict__ S,
    int Lq, int Lkv, int inner, int kvstride, int heads, int dh, float scale, int t_base)
{
  const int lane = threadIdx.x;
  const int n0 = blockIdx.x * 16;   // key tile
  const int m0 = blockIdx.y * 16;   // query tile
  const int z  = blockIdx.z;
  const int t = t_base + z;
  const int gb = t / heads, h = t % heads;
  const float* Qb = Q + (size_t)gb * Lq * inner + h * dh;
  const float* Kb = KV + (size_t)gb * Lkv * kvstride + h * dh;
  const int half = lane >> 4, mr = lane & 15;
  v8f acc = {};
  for (int k0 = 0; k0 < dh; k0 += 32) {
    v16us au, bu;
    const float* qrow = Qb + (size_t)(m0 + mr) * inner;
#pragma unroll
    for (int j = 0; j < 8; ++j) {
      int kb = k0 + half * 16 + 2 * j;
      au[2 * j]     = f2bf(qrow[kb]);
      au[2 * j + 1] = f2bf(qrow[kb + 1]);
    }
    const float* krow = Kb + (size_t)(n0 + mr) * kvstride;
#pragma unroll
    for (int e = 0; e < 16; ++e)
      bu[e] = f2bf(krow[k0 + half * 16 + e]);
    acc = __builtin_amdgcn_wmma_f32_16x16x32_bf16(false, __builtin_bit_cast(v16bf, au),
                                                  false, __builtin_bit_cast(v16bf, bu),
                                                  (short)0, acc, false, false);
  }
  float* Sb = S + (size_t)z * Lq * Lkv;
#pragma unroll
  for (int r = 0; r < 8; ++r)
    Sb[(size_t)(m0 + half * 8 + r) * Lkv + n0 + mr] = acc[r] * scale;
}

// ---------------- row softmax: P(bf16) = softmax(S) ----------------
__global__ void softmax_kernel(const float* __restrict__ S, unsigned short* __restrict__ P,
                               int Lq, int Lkv)
{
  __shared__ float red[8];
  const int row = blockIdx.x, z = blockIdx.y;
  const float* sr = S + ((size_t)z * Lq + row) * Lkv;
  unsigned short* pr = P + ((size_t)z * Lq + row) * Lkv;
  const int tid = threadIdx.x, nt = blockDim.x;
  const int lane = tid & 31, wid = tid >> 5;
  const int nw = (nt + 31) >> 5;
  float m = -3.4e38f;
  for (int i = tid; i < Lkv; i += nt) m = fmaxf(m, sr[i]);
#pragma unroll
  for (int o = 16; o >= 1; o >>= 1) m = fmaxf(m, __shfl_xor(m, o, 32));
  if (lane == 0) red[wid] = m;
  __syncthreads();
  if (tid == 0) { float v = red[0]; for (int w = 1; w < nw; ++w) v = fmaxf(v, red[w]); red[0] = v; }
  __syncthreads();
  m = red[0];
  __syncthreads();
  float ssum = 0.f;
  for (int i = tid; i < Lkv; i += nt) ssum += __expf(sr[i] - m);
#pragma unroll
  for (int o = 16; o >= 1; o >>= 1) ssum += __shfl_xor(ssum, o, 32);
  if (lane == 0) red[wid] = ssum;
  __syncthreads();
  if (tid == 0) { float v = 0.f; for (int w = 0; w < nw; ++w) v += red[w]; red[0] = v; }
  __syncthreads();
  const float inv = 1.0f / red[0];
  for (int i = tid; i < Lkv; i += nt) pr[i] = f2bf(__expf(sr[i] - m) * inv);
}

// ---------------- O[gb,Lq,inner](bf16) = head slice of P(bf16) @ V(f32) ----------------
__global__ __launch_bounds__(32) void attn_out_kernel(
    const unsigned short* __restrict__ P, const float* __restrict__ KV,
    unsigned short* __restrict__ O,
    int Lq, int Lkv, int inner, int kvstride, int heads, int dh, int v_off, int t_base)
{
  const int lane = threadIdx.x;
  const int n0 = blockIdx.x * 16;   // within dh
  const int m0 = blockIdx.y * 16;   // query tile
  const int z = blockIdx.z;
  const int t = t_base + z;
  const int gb = t / heads, h = t % heads;
  const unsigned short* Pb = P + (size_t)z * Lq * Lkv;
  const float* Vb = KV + (size_t)gb * Lkv * kvstride + v_off + h * dh;
  const int half = lane >> 4, mr = lane & 15;
  v8f acc = {};
  for (int k0 = 0; k0 < Lkv; k0 += 32) {
    v16us au, bu;
    const unsigned short* prow = Pb + (size_t)(m0 + mr) * Lkv;
#pragma unroll
    for (int j = 0; j < 8; ++j) {
      int kb = k0 + half * 16 + 2 * j;
      unsigned pair = *(const unsigned*)(prow + kb);
      au[2 * j]     = (unsigned short)(pair & 0xFFFFu);
      au[2 * j + 1] = (unsigned short)(pair >> 16);
    }
#pragma unroll
    for (int e = 0; e < 16; ++e)
      bu[e] = f2bf(Vb[(size_t)(k0 + half * 16 + e) * kvstride + n0 + mr]);
    acc = __builtin_amdgcn_wmma_f32_16x16x32_bf16(false, __builtin_bit_cast(v16bf, au),
                                                  false, __builtin_bit_cast(v16bf, bu),
                                                  (short)0, acc, false, false);
  }
#pragma unroll
  for (int r = 0; r < 8; ++r)
    O[((size_t)gb * Lq + m0 + half * 8 + r) * inner + h * dh + n0 + mr] = f2bf(acc[r]);
}

// ---------------- LayerNorm -> bf16 (optional [B,G,L,d]->[G,B,L,d] permute, per-group weights) ----------------
__global__ void ln_kernel(const float* __restrict__ X, unsigned short* __restrict__ Y,
                          const float* __restrict__ Gw, const float* __restrict__ Bw,
                          int L, int d, int wper, int permB, int permG)
{
  __shared__ float red[8];
  const int r = blockIdx.x;
  const int s = r / L, ii = r % L;
  const int widx = s / wper;
  size_t inrow;
  if (permB > 0) {
    int g = s / permB, b = s % permB;
    inrow = (size_t)(b * permG + g) * L + ii;
  } else inrow = (size_t)r;
  const float* x = X + inrow * (size_t)d;
  const int tid = threadIdx.x, nt = blockDim.x;
  const int lane = tid & 31, wid = tid >> 5;
  const int nw = (nt + 31) >> 5;
  float sum = 0.f;
  for (int c = tid; c < d; c += nt) sum += x[c];
#pragma unroll
  for (int o = 16; o >= 1; o >>= 1) sum += __shfl_xor(sum, o, 32);
  if (lane == 0) red[wid] = sum;
  __syncthreads();
  if (tid == 0) { float v = 0.f; for (int w = 0; w < nw; ++w) v += red[w]; red[0] = v; }
  __syncthreads();
  const float mean = red[0] / d;
  __syncthreads();
  float vs = 0.f;
  for (int c = tid; c < d; c += nt) { float tv = x[c] - mean; vs += tv * tv; }
#pragma unroll
  for (int o = 16; o >= 1; o >>= 1) vs += __shfl_xor(vs, o, 32);
  if (lane == 0) red[wid] = vs;
  __syncthreads();
  if (tid == 0) { float v = 0.f; for (int w = 0; w < nw; ++w) v += red[w]; red[0] = v; }
  __syncthreads();
  const float rstd = rsqrtf(red[0] / d + 1e-5f);
  for (int c = tid; c < d; c += nt)
    Y[(size_t)r * d + c] = f2bf((x[c] - mean) * rstd * Gw[(size_t)widx * d + c]
                                + Bw[(size_t)widx * d + c]);
}

// ---------------- GEGLU: out(bf16)[row,4d] = a * gelu_exact(g), h = [a|g] ----------------
__global__ void geglu_kernel(const float* __restrict__ H, unsigned short* __restrict__ O,
                             int d4, int total)
{
  int idx = blockIdx.x * blockDim.x + threadIdx.x;
  if (idx >= total) return;
  int row = idx / d4, c = idx % d4;
  const float* hr = H + (size_t)row * (2 * d4);
  float a = hr[c], g = hr[d4 + c];
  float ge = 0.5f * g * (1.0f + erff(g * 0.70710678118654752f));
  O[idx] = f2bf(a * ge);
}

__global__ void cvt_bf16_kernel(const float* __restrict__ X, unsigned short* __restrict__ Y, int n) {
  int i = blockIdx.x * blockDim.x + threadIdx.x;
  if (i < n) Y[i] = f2bf(X[i]);
}

__global__ void add_kernel(const float* __restrict__ A, const float* __restrict__ B,
                           float* __restrict__ C, int n) {
  int i = blockIdx.x * blockDim.x + threadIdx.x;
  if (i < n) C[i] = A[i] + B[i];
}

// z[G,B,L,D] -> out[B, G*L, D]
__global__ void unperm_kernel(const float* __restrict__ Z, float* __restrict__ O,
                              int G, int B, int L, int D, int total)
{
  int o = blockIdx.x * blockDim.x + threadIdx.x;
  if (o >= total) return;
  int c = o % D, r = o / D;
  int ii = r % L, t = r / L;
  int g = t % G, b = t / G;
  O[o] = Z[((size_t)(g * B + b) * L + ii) * D + c];
}

// ---------------- host-side parameter tree (jax pytree flatten order: sorted dict keys) ----------------
struct FFP  { const float *b1, *b2, *w1, *w2, *lnb, *lng; };
struct AttP { const float *bo, *wkv, *wo, *wq; };
struct SelfP { AttP att; FFP ff[2]; const float *lnb, *lng; };
struct LayerP {
  AttP cross; FFP cff[2];
  const float *clnc_b, *clnc_g, *clnq_b, *clnq_g;
  SelfP self[18];
  const float* ind;
};

extern "C" void kernel_launch(void* const* d_in, const int* in_sizes, int n_in,
                              void* d_out, int out_size, void* d_ws, size_t ws_size,
                              hipStream_t stream)
{
  static const int GROUPS_[7] = {16, 4, 1, 1, 1, 4, 16};
  static const int SA_[7]     = {2, 2, 18, 2, 1, 1, 1};
  static const int HEADS_[7]  = {4, 8, 16, 32, 16, 8, 4};
  static const int LCH_[7]    = {128, 256, 512, 1024, 512, 256, 128};
  static const int LVPG_[7]   = {128, 256, 256, 64, 256, 256, 128};
  static const int ICH_[7]    = {32, 128, 256, 512, 1024, 512, 256};
  const int BB = 2, NN = 2048;
  (void)out_size; (void)ws_size;

  // x may be first (dict insertion order) or last (full sorted tree-flatten)
  int xpos, idx;
  if (in_sizes[0] == BB * NN * 32) { xpos = 0; idx = 1; }
  else                              { xpos = n_in - 1; idx = 0; }
  auto F = [&]() { return (const float*)d_in[idx++]; };

  AttP fin; fin.bo = F(); fin.wkv = F(); fin.wo = F(); fin.wq = F();
  static LayerP LP[7];
  for (int i = 0; i < 7; ++i) {
    LayerP& p = LP[i];
    p.cross.bo = F(); p.cross.wkv = F(); p.cross.wo = F(); p.cross.wq = F();
    for (int f = 0; f < 2; ++f) {
      FFP& q = p.cff[f];
      q.b1 = F(); q.b2 = F(); q.w1 = F(); q.w2 = F(); q.lnb = F(); q.lng = F();
    }
    p.clnc_b = F(); p.clnc_g = F(); p.clnq_b = F(); p.clnq_g = F();
    for (int s = 0; s < SA_[i]; ++s) {
      SelfP& sp = p.self[s];
      sp.att.bo = F(); sp.att.wkv = F(); sp.att.wo = F(); sp.att.wq = F();
      for (int f = 0; f < 2; ++f) {
        FFP& q = sp.ff[f];
        q.b1 = F(); q.b2 = F(); q.w1 = F(); q.w2 = F(); q.lnb = F(); q.lng = F();
      }
      sp.lnb = F(); sp.lng = F();
    }
    p.ind = F();
  }
  const float* xin0 = (const float*)d_in[xpos];

  // ---------------- workspace ----------------
  char* wp = (char*)d_ws;
  auto alloc = [&](size_t bytes) -> void* {
    void* r = (void*)wp; wp += (bytes + 255) & ~(size_t)255; return r;
  };
  float* xA = (float*)alloc((size_t)524288 * 4);
  float* xB = (float*)alloc((size_t)524288 * 4);
  float* xC = (float*)alloc((size_t)524288 * 4);
  float* cacheA = (float*)alloc((size_t)524288 * 4);
  float* cacheB = (float*)alloc((size_t)262144 * 4);
  unsigned short* bfA = (unsigned short*)alloc((size_t)524288 * 2);
  unsigned short* bfB = (unsigned short*)alloc((size_t)262144 * 2);
  float* qb  = (float*)alloc((size_t)2097152 * 4);
  float* kvb = (float*)alloc((size_t)4194304 * 4);
  float* sc  = (float*)alloc((size_t)8388608 * 4);
  unsigned short* pb   = (unsigned short*)alloc((size_t)8388608 * 2);
  unsigned short* aout = (unsigned short*)alloc((size_t)2097152 * 2);
  float* zb  = (float*)alloc((size_t)524288 * 4);
  float* ffh = (float*)alloc((size_t)4194304 * 4);
  unsigned short* ffg = (unsigned short*)alloc((size_t)2097152 * 2);
  unsigned short* wtb = (unsigned short*)alloc((size_t)8388608 * 2);  // max weight: 1024x8192

  // ---------------- launch helpers ----------------
  auto gemm = [&](const unsigned short* A, const float* W, const float* bias, float* Cc,
                  int M, int N, int K, int Ap, int Wp, int S) {
    int WS = S / Wp;  // distinct weight slices
    wt_kernel<<<dim3(N / 16, K / 16, WS), dim3(16, 16), 0, stream>>>(W, wtb, K, N);
    gemm_bf16_kernel<<<dim3(N / 16, M / 64, S), dim3(32), 0, stream>>>(
        A, wtb, bias, Cc, M, N, K, Ap, Wp);
  };
  auto lnk = [&](const float* X, unsigned short* Y, const float* g, const float* b,
                 int rows, int L, int d, int wper, int permB, int permG) {
    ln_kernel<<<dim3(rows), dim3(imin_(256, d)), 0, stream>>>(X, Y, g, b, L, d, wper, permB, permG);
  };
  auto attn_core = [&](const float* Q, const float* KV, unsigned short* Ob,
                       int Lq, int Lkv, int inner, int H, int T) {
    int dh = inner / H;
    float scale = 1.0f / sqrtf((float)dh);
    attn_scores_kernel<<<dim3(Lkv / 16, Lq / 16, T), dim3(32), 0, stream>>>(
        Q, KV, sc, Lq, Lkv, inner, 2 * inner, H, dh, scale, 0);
    softmax_kernel<<<dim3(Lq, T), dim3(256), 0, stream>>>(sc, pb, Lq, Lkv);
    attn_out_kernel<<<dim3(dh / 16, Lq / 16, T), dim3(32), 0, stream>>>(
        pb, KV, Ob, Lq, Lkv, inner, 2 * inner, H, dh, inner, 0);
  };
  auto ff_block = [&](float* z, const FFP& f, int GB, int L, int D) {
    lnk(z, bfA, f.lng, f.lnb, GB * L, L, D, BB, 0, 0);
    gemm(bfA, f.w1, f.b1, ffh, L, 8 * D, D, 1, BB, GB);
    int tot = GB * L * 4 * D;
    geglu_kernel<<<dim3((tot + 255) / 256), dim3(256), 0, stream>>>(ffh, ffg, 4 * D, tot);
    gemm(ffg, f.w2, f.b2, z, L, D, 4 * D, 1, BB, GB);
  };

  // ---------------- 7 HPC layers ----------------
  const float* cur = xin0;
  int curlen = NN;
  for (int i = 0; i < 7; ++i) {
    const LayerP& p = LP[i];
    const int G = GROUPS_[i], L = LVPG_[i], D = LCH_[i], Ci = ICH_[i], H = HEADS_[i];
    const int nc = curlen / G;
    const int GB = G * BB;

    const float* inp = cur;
    if (i == 5) {  // decode 'b' (layer2 output)
      int n = BB * curlen * Ci;
      add_kernel<<<dim3((n + 255) / 256), dim3(256), 0, stream>>>(cur, cacheB, xC, n);
      inp = xC;
    } else if (i == 6) {  // decode 'a' (layer1 output)
      int n = BB * curlen * Ci;
      add_kernel<<<dim3((n + 255) / 256), dim3(256), 0, stream>>>(cur, cacheA, xC, n);
      inp = xC;
    }

    // ---- cross attention: latents -> chunk ----
    lnk(inp, bfA, p.clnc_g, p.clnc_b, GB * nc, nc, Ci, BB, BB, G);  // ctx LN (permuted read)
    lnk(p.ind, bfB, p.clnq_g, p.clnq_b, G * L, L, D, 1, 0, 0);      // latent LN
    gemm(bfB, p.cross.wq, nullptr, qb, L, D, D, BB, BB, GB);        // q (A broadcast over batch)
    gemm(bfA, p.cross.wkv, nullptr, kvb, nc, 2 * D, Ci, 1, BB, GB); // kv
    attn_core(qb, kvb, aout, L, nc, D, H, GB * H);
    gemm(aout, p.cross.wo, p.cross.bo, zb, L, D, D, 1, BB, GB);     // out proj
    for (int f = 0; f < 2; ++f) ff_block(zb, p.cff[f], GB, L, D);

    // ---- self attention stack ----
    for (int s = 0; s < SA_[i]; ++s) {
      const SelfP& sp = p.self[s];
      lnk(zb, bfA, sp.lng, sp.lnb, GB * L, L, D, BB, 0, 0);
      gemm(bfA, sp.att.wq, nullptr, qb, L, D, D, 1, BB, GB);
      gemm(bfA, sp.att.wkv, nullptr, kvb, L, 2 * D, D, 1, BB, GB);
      attn_core(qb, kvb, aout, L, L, D, H, GB * H);
      gemm(aout, sp.att.wo, sp.att.bo, zb, L, D, D, 1, BB, GB);
      for (int f = 0; f < 2; ++f) ff_block(zb, sp.ff[f], GB, L, D);
    }

    // ---- un-permute [G,B,L,D] -> [B, G*L, D]; encode layers land in cache ----
    float* dest = (i == 1) ? cacheA : (i == 2) ? cacheB : ((i % 2 == 0) ? xA : xB);
    int tot = G * BB * L * D;
    unperm_kernel<<<dim3((tot + 255) / 256), dim3(256), 0, stream>>>(zb, dest, G, BB, L, D, tot);
    cur = dest;
    curlen = G * L;
  }

  // ---------------- final attention: q from original x, kv from layer-6 output ----------------
  {
    int nq = BB * NN * 32;
    cvt_bf16_kernel<<<dim3((nq + 255) / 256), dim3(256), 0, stream>>>(xin0, bfB, nq);
    int nk = BB * NN * 128;
    cvt_bf16_kernel<<<dim3((nk + 255) / 256), dim3(256), 0, stream>>>(cur, bfA, nk);
    gemm(bfB, fin.wq, nullptr, qb, NN, 512, 32, 1, BB, BB);     // [2,2048,512]
    gemm(bfA, fin.wkv, nullptr, kvb, NN, 1024, 128, 1, BB, BB); // [2,2048,1024]
    const int H = 8, dh = 64;
    const float scale = 0.125f;  // 64^-0.5
    for (int t0 = 0; t0 < BB * H; t0 += 2) {  // 2 (b,h)-slices per pass: 33.5MB score scratch
      attn_scores_kernel<<<dim3(NN / 16, NN / 16, 2), dim3(32), 0, stream>>>(
          qb, kvb, sc, NN, NN, 512, 1024, H, dh, scale, t0);
      softmax_kernel<<<dim3(NN, 2), dim3(256), 0, stream>>>(sc, pb, NN, NN);
      attn_out_kernel<<<dim3(dh / 16, NN / 16, 2), dim3(32), 0, stream>>>(
          pb, kvb, aout, NN, NN, 512, 1024, H, dh, 512, t0);
    }
    gemm(aout, fin.wo, fin.bo, (float*)d_out, NN, 128, 512, 1, BB, BB);
  }
}
